// NanoporeVQModel_15101105013441
// MI455X (gfx1250) — matrix operand order repaired and checked by reading, compiled, not run on hardware
//
#include <hip/hip_runtime.h>
#include <hip/hip_bf16.h>

#define BB   8
#define TT   10240
#define LL   2048
#define DD   64
#define KK   8192
#define T2   10236   // decoder length: 5*L - 4

#define NT   128     // codes per LDS-staged tile
#define ROWB 144     // padded LDS row stride (128B row + 16B pad -> bank-conflict free)

typedef __bf16 bf16;
typedef __attribute__((ext_vector_type(16))) __bf16 v16bf;
typedef __attribute__((ext_vector_type(8)))  float  v8f;
typedef __attribute__((ext_vector_type(4)))  unsigned int u32x4;
typedef __attribute__((ext_vector_type(8)))  int  i32x8;
typedef __attribute__((ext_vector_type(4)))  int  i32x4;

__device__ __forceinline__ float silu_f(float x) {
    return x * (1.0f / (1.0f + __expf(-x)));
}

// ---------------- codebook prep: f32 -> bf16 + squared norms ----------------
__global__ void prep_codebook(const float* __restrict__ cb,
                              bf16* __restrict__ cbb,
                              float* __restrict__ cnorm) {
    int n = blockIdx.x * blockDim.x + threadIdx.x;
    if (n >= KK) return;
    const float* r = cb + (size_t)n * DD;
    bf16* o = cbb + (size_t)n * DD;
    float s = 0.f;
    for (int i = 0; i < DD; ++i) {
        float v = r[i];
        s += v * v;
        o[i] = (bf16)v;
    }
    cnorm[n] = s;
}

// ---------------- encoder conv1: 1->16, k5, p2, + silu ----------------
__global__ void enc_conv1(const float* __restrict__ x, const float* __restrict__ w,
                          const float* __restrict__ bia, float* __restrict__ h1) {
    int tid = blockIdx.x * blockDim.x + threadIdx.x;
    if (tid >= BB * TT) return;
    int t = tid % TT, b = tid / TT;
    float xin[5];
    #pragma unroll
    for (int k = 0; k < 5; ++k) {
        int tt = t - 2 + k;
        xin[k] = (tt >= 0 && tt < TT) ? x[(size_t)b * TT + tt] : 0.f;
    }
    #pragma unroll
    for (int c = 0; c < 16; ++c) {
        float acc = bia[c];
        #pragma unroll
        for (int k = 0; k < 5; ++k) acc += w[c * 5 + k] * xin[k];
        h1[((size_t)b * 16 + c) * TT + t] = silu_f(acc);
    }
}

// ---------------- BN stats over (B,C,T) layout -> ab[c]=scale, ab[C+c]=shift --
__global__ void bn_stats_nct(const float* __restrict__ h, const float* __restrict__ g,
                             const float* __restrict__ bb, float* __restrict__ ab,
                             int C, int Tlen) {
    __shared__ float s1[256], s2[256];
    int c = blockIdx.x;
    long n = (long)BB * Tlen;
    float sum = 0.f, sq = 0.f;
    for (long i = threadIdx.x; i < n; i += blockDim.x) {
        long b = i / Tlen, t = i - b * Tlen;
        float v = h[((size_t)b * C + c) * Tlen + t];
        sum += v; sq += v * v;
    }
    s1[threadIdx.x] = sum; s2[threadIdx.x] = sq;
    __syncthreads();
    for (int off = 128; off > 0; off >>= 1) {
        if ((int)threadIdx.x < off) {
            s1[threadIdx.x] += s1[threadIdx.x + off];
            s2[threadIdx.x] += s2[threadIdx.x + off];
        }
        __syncthreads();
    }
    if (threadIdx.x == 0) {
        float inv = 1.0f / (float)n;
        float mu = s1[0] * inv;
        float var = s2[0] * inv - mu * mu;
        float a = g[c] * rsqrtf(var + 1e-5f);
        ab[c] = a;
        ab[C + c] = bb[c] - mu * a;
    }
}

__global__ void bn_apply_nct(float* __restrict__ h, const float* __restrict__ ab,
                             int C, int Tlen) {
    long total = (long)BB * C * Tlen;
    for (long e = blockIdx.x * (long)blockDim.x + threadIdx.x; e < total;
         e += (long)gridDim.x * blockDim.x) {
        int c = (int)((e / Tlen) % C);
        h[e] = h[e] * ab[c] + ab[C + c];
    }
}

// ---------------- BN over (rows, C) row-major layout ----------------
__global__ void bn_stats_rc(const float* __restrict__ z, const float* __restrict__ g,
                            const float* __restrict__ bb, float* __restrict__ ab,
                            int C, int rows) {
    __shared__ float s1[256], s2[256];
    int c = blockIdx.x;
    float sum = 0.f, sq = 0.f;
    for (int i = threadIdx.x; i < rows; i += blockDim.x) {
        float v = z[(size_t)i * C + c];
        sum += v; sq += v * v;
    }
    s1[threadIdx.x] = sum; s2[threadIdx.x] = sq;
    __syncthreads();
    for (int off = 128; off > 0; off >>= 1) {
        if ((int)threadIdx.x < off) {
            s1[threadIdx.x] += s1[threadIdx.x + off];
            s2[threadIdx.x] += s2[threadIdx.x + off];
        }
        __syncthreads();
    }
    if (threadIdx.x == 0) {
        float inv = 1.0f / (float)rows;
        float mu = s1[0] * inv;
        float var = s2[0] * inv - mu * mu;
        float a = g[c] * rsqrtf(var + 1e-5f);
        ab[c] = a;
        ab[C + c] = bb[c] - mu * a;
    }
}

__global__ void bn_apply_rc(float* __restrict__ z, const float* __restrict__ ab,
                            int C, int rows) {
    long total = (long)rows * C;
    for (long e = blockIdx.x * (long)blockDim.x + threadIdx.x; e < total;
         e += (long)gridDim.x * blockDim.x) {
        int c = (int)(e % C);
        z[e] = z[e] * ab[c] + ab[C + c];
    }
}

// ---------------- encoder conv2: 16->32, k5, p2, + silu ----------------
__global__ void enc_conv2(const float* __restrict__ h1, const float* __restrict__ w,
                          const float* __restrict__ bia, float* __restrict__ h2) {
    long tid = blockIdx.x * (long)blockDim.x + threadIdx.x;
    if (tid >= (long)BB * 32 * TT) return;
    int t = (int)(tid % TT);
    int co = (int)((tid / TT) % 32);
    int b = (int)(tid / ((long)TT * 32));
    float acc = bia[co];
    for (int ci = 0; ci < 16; ++ci) {
        const float* hp = h1 + ((size_t)b * 16 + ci) * TT;
        const float* wp = w + (co * 16 + ci) * 5;
        #pragma unroll
        for (int k = 0; k < 5; ++k) {
            int tt = t - 2 + k;
            if (tt >= 0 && tt < TT) acc += hp[tt] * wp[k];
        }
    }
    h2[tid] = silu_f(acc);
}

// ---------------- encoder conv3: 32->64, k25, s5, p12, + tanh ----------------
// output stored row-major: z[(b*L + l)*64 + c]
__global__ void enc_conv3(const float* __restrict__ h2, const float* __restrict__ w,
                          const float* __restrict__ bia, float* __restrict__ z) {
    long tid = blockIdx.x * (long)blockDim.x + threadIdx.x;
    if (tid >= (long)BB * LL * DD) return;
    int c = (int)(tid % DD);
    int l = (int)((tid / DD) % LL);
    int b = (int)(tid / ((long)DD * LL));
    float acc = bia[c];
    for (int ci = 0; ci < 32; ++ci) {
        const float* hp = h2 + ((size_t)b * 32 + ci) * TT;
        const float* wp = w + (c * 32 + ci) * 25;
        #pragma unroll
        for (int k = 0; k < 25; ++k) {
            int tt = 5 * l - 12 + k;
            if (tt >= 0 && tt < TT) acc += hp[tt] * wp[k];
        }
    }
    z[tid] = tanhf(acc);
}

// ---------------- VQ argmin: WMMA bf16 + TDM codebook staging in LDS ---------
// 8 waves/block; each wave owns a 16-row zf tile. The block cooperatively
// stages 128-code tiles of the bf16 codebook into LDS (double-buffered) via the
// Tensor Data Mover (wave 0 issues, TENSORcnt waits), with LDS row padding
// (128B row + 16B pad) for conflict-free ds reads of B fragments.
__global__ __launch_bounds__(256) void vq_argmin(const float* __restrict__ zf,
                                                 const bf16* __restrict__ cbb,
                                                 const float* __restrict__ cnorm,
                                                 int* __restrict__ idx_i,
                                                 float* __restrict__ idx_f) {
    __shared__ __align__(16) char scb[2][NT * ROWB];
    __shared__ float svals[8][16][16];
    __shared__ int   sids[8][16][16];

    const int wave = threadIdx.x >> 5;
    const int lane = threadIdx.x & 31;
    const int m    = lane & 15;
    const int hi   = lane >> 4;                 // lane half: K-split per ISA layout
    const int row0 = (blockIdx.x * 8 + wave) * 16;

    // A tile: 16 rows x K=64 (two 16x32 bf16 fragments).
    // A k order used here (contiguous): lane half 0 holds K={0..15}, half 1 holds K={16..31}.
    v16bf a0, a1;
    {
        const float* zr = zf + (size_t)(row0 + m) * DD;
        #pragma unroll
        for (int i = 0; i < 16; ++i) {
            int r = i >> 1, s = i & 1;
            int k = hi * 16 + i;
            a0[i] = (bf16)zr[k];
            a1[i] = (bf16)zr[32 + k];
        }
    }

    float bestv[8];
    int   besti[8];
    #pragma unroll
    for (int r = 0; r < 8; ++r) { bestv[r] = 3.4e38f; besti[r] = 0; }

#if __has_builtin(__builtin_amdgcn_tensor_load_to_lds)
    // TDM descriptor: 2D tensor, data_size=4B, row = 32 units (128B of bf16),
    // tile = 32 x NT, pad_enable: 4 DWORDs after every 32 DWORDs (row stride 144B).
    auto issue_tile = [&](int jt, int bufi) {
        unsigned long long ga = (unsigned long long)(const void*)cbb
                              + (unsigned long long)jt * (NT * DD * 2);
        unsigned lds_off = (unsigned)(unsigned long long)(const void*)&scb[bufi][0];
        u32x4 g0 = { 1u,                                  // count=1 (valid D#)
                     lds_off,                             // lds_addr
                     (unsigned)(ga & 0xffffffffu),
                     (unsigned)((ga >> 32) & 0x1ffffffu) | (2u << 30) }; // type=2
        i32x8 g1 = { (int)((2u << 16) | (1u << 20) | (4u << 22) | (3u << 25)),
                     // ^ data_size=4B, pad_enable, pad_interval=32dw, pad_amount=4dw
                     (int)(32u << 16),                    // tensor_dim0 = 32 units
                     (int)(8192u << 16),                  // tensor_dim1 = 8192 rows
                     (int)(32u << 16),                    // tile_dim0 = 32 units
                     (int)NT,                             // tile_dim1 = 128 rows
                     32,                                  // tensor_dim0_stride = 32
                     0, 0 };
        i32x4 g2 = { 0, 0, 0, 0 };
        i32x4 g3 = { 0, 0, 0, 0 };
#if defined(__clang_major__) && __clang_major__ >= 23
        i32x8 gx = { 0, 0, 0, 0, 0, 0, 0, 0 };
        __builtin_amdgcn_tensor_load_to_lds(g0, g1, g2, g3, gx, 0);
#else
        __builtin_amdgcn_tensor_load_to_lds(g0, g1, g2, g3, 0);
#endif
    };

    const int NTILES = KK / NT;
    const int kb = hi * 16;                     // B layout: lane halves split K 16/16
    if (wave == 0) {
        issue_tile(0, 0);
        __builtin_amdgcn_s_wait_tensorcnt(0);
    }
    __syncthreads();

    for (int jt = 0; jt < NTILES; ++jt) {
        const int bufi = jt & 1;
        if (wave == 0 && (jt + 1) < NTILES) issue_tile(jt + 1, bufi ^ 1);
        const char* tile = &scb[bufi][0];
        #pragma unroll
        for (int j2 = 0; j2 < NT / 16; ++j2) {
            const int nl = j2 * 16 + m;
            const int n  = jt * NT + nl;
            const bf16* cr = (const bf16*)(tile + nl * ROWB);
            v16bf b0, b1;
            #pragma unroll
            for (int i = 0; i < 16; ++i) {
                b0[i] = cr[kb + i];
                b1[i] = cr[32 + kb + i];
            }
            v8f c = {};
            c = __builtin_amdgcn_wmma_f32_16x16x32_bf16(false, a0, false, b0,
                                                        (short)0, c, false, false);
            c = __builtin_amdgcn_wmma_f32_16x16x32_bf16(false, a1, false, b1,
                                                        (short)0, c, false, false);
            const float cn = cnorm[n];
            #pragma unroll
            for (int r = 0; r < 8; ++r) {
                float sc = cn - 2.0f * c[r];    // D layout: vgpr r -> M=r+8*hi
                if (sc < bestv[r]) { bestv[r] = sc; besti[r] = n; }
            }
        }
        if (wave == 0 && (jt + 1) < NTILES) __builtin_amdgcn_s_wait_tensorcnt(0);
        __syncthreads();
    }
#else
    // Fallback: stream B tiles straight from global memory.
    const int kb = hi * 16;
    for (int j = 0; j < KK / 16; ++j) {
        const int n = j * 16 + m;
        const bf16* cr = cbb + (size_t)n * DD;
        v16bf b0, b1;
        #pragma unroll
        for (int i = 0; i < 16; ++i) {
            b0[i] = cr[kb + i];
            b1[i] = cr[32 + kb + i];
        }
        v8f c = {};
        c = __builtin_amdgcn_wmma_f32_16x16x32_bf16(false, a0, false, b0,
                                                    (short)0, c, false, false);
        c = __builtin_amdgcn_wmma_f32_16x16x32_bf16(false, a1, false, b1,
                                                    (short)0, c, false, false);
        const float cn = cnorm[n];
        #pragma unroll
        for (int r = 0; r < 8; ++r) {
            float sc = cn - 2.0f * c[r];
            if (sc < bestv[r]) { bestv[r] = sc; besti[r] = n; }
        }
    }
#endif

    #pragma unroll
    for (int r = 0; r < 8; ++r) {
        int mm = r + hi * 8;
        svals[wave][mm][m] = bestv[r];
        sids[wave][mm][m]  = besti[r];
    }
    __syncthreads();
    if (lane < 16) {
        float bv = svals[wave][lane][0];
        int   bi = sids[wave][lane][0];
        #pragma unroll
        for (int n2 = 1; n2 < 16; ++n2) {
            float v = svals[wave][lane][n2];
            if (v < bv) { bv = v; bi = sids[wave][lane][n2]; }
        }
        int row = row0 + lane;
        idx_i[row] = bi;
        idx_f[row] = (float)bi;
    }
}

// ---------------- commit loss ----------------
__global__ void loss_zero(float* acc) {
    if (blockIdx.x == 0 && threadIdx.x == 0) *acc = 0.f;
}

__global__ void loss_acc_k(const float* __restrict__ z, const float* __restrict__ cb,
                           const int* __restrict__ idx, float* __restrict__ acc) {
    int row = blockIdx.x * blockDim.x + threadIdx.x;
    if (row >= BB * LL) return;
    const float* zr = z + (size_t)row * DD;
    const float* cr = cb + (size_t)idx[row] * DD;
    float s = 0.f;
    for (int i = 0; i < DD; ++i) {
        float d = cr[i] - zr[i];
        s += d * d;
    }
    atomicAdd(acc, s);
}

__global__ void loss_fin(const float* __restrict__ acc, float* __restrict__ out) {
    if (blockIdx.x == 0 && threadIdx.x == 0)
        out[0] = 2.0f * acc[0] / (float)((long)BB * LL * DD);
}

// ---------------- gather q = codebook[idx] (straight-through fwd value) -------
__global__ void gather_q(const float* __restrict__ cb, const int* __restrict__ idx,
                         float* __restrict__ q) {
    long tid = blockIdx.x * (long)blockDim.x + threadIdx.x;
    if (tid >= (long)BB * LL * DD) return;
    long row = tid / DD;
    int  d   = (int)(tid % DD);
    q[tid] = cb[(size_t)idx[row] * DD + d];
}

// ---------------- decoder transposed conv: 64->64, k25, lhs_dilation=5, p12 ---
// wconv[o,i,k] = dec_wt[i,o,24-k]; q is (b*L+l, 64) row-major. + silu
__global__ void dec_tconv1(const float* __restrict__ q, const float* __restrict__ wt,
                           float* __restrict__ hd1) {
    long tid = blockIdx.x * (long)blockDim.x + threadIdx.x;
    if (tid >= (long)BB * DD * T2) return;
    int t  = (int)(tid % T2);
    int co = (int)((tid / T2) % DD);
    int b  = (int)(tid / ((long)T2 * DD));
    float acc = 0.f;
    #pragma unroll
    for (int k = 0; k < 25; ++k) {
        int v = t + k - 12;
        if (v >= 0 && (v % 5) == 0) {
            int l = v / 5;
            if (l < LL) {
                const float* qr = q + ((size_t)b * LL + l) * DD;
                const float* wp = wt + co * 25 + (24 - k);   // wt[i*64*25 + co*25 + 24-k]
                float s = 0.f;
                for (int i = 0; i < DD; ++i) s += qr[i] * wp[(size_t)i * DD * 25];
                acc += s;
            }
        }
    }
    hd1[tid] = silu_f(acc);
}

// ---------------- decoder conv2: 64->64, k5, p2, + silu ----------------
__global__ void dec_conv2(const float* __restrict__ hd1, const float* __restrict__ w,
                          const float* __restrict__ bia, float* __restrict__ hd2) {
    long tid = blockIdx.x * (long)blockDim.x + threadIdx.x;
    if (tid >= (long)BB * DD * T2) return;
    int t  = (int)(tid % T2);
    int co = (int)((tid / T2) % DD);
    int b  = (int)(tid / ((long)T2 * DD));
    float acc = bia[co];
    for (int ci = 0; ci < DD; ++ci) {
        const float* hp = hd1 + ((size_t)b * DD + ci) * T2;
        const float* wp = w + (co * DD + ci) * 5;
        #pragma unroll
        for (int k = 0; k < 5; ++k) {
            int tt = t - 2 + k;
            if (tt >= 0 && tt < T2) acc += hp[tt] * wp[k];
        }
    }
    hd2[tid] = silu_f(acc);
}

// ---------------- decoder conv3: 64->1, k1, pad to T ----------------
__global__ void dec_conv3(const float* __restrict__ hd2, const float* __restrict__ w3,
                          const float* __restrict__ b3, float* __restrict__ recon) {
    int tid = blockIdx.x * blockDim.x + threadIdx.x;
    if (tid >= BB * TT) return;
    int t = tid % TT, b = tid / TT;
    float acc = 0.f;
    if (t < T2) {
        acc = b3[0];
        for (int i = 0; i < DD; ++i)
            acc += hd2[((size_t)b * DD + i) * T2 + t] * w3[i];
    }
    recon[tid] = acc;
}

extern "C" void kernel_launch(void* const* d_in, const int* in_sizes, int n_in,
                              void* d_out, int out_size, void* d_ws, size_t ws_size,
                              hipStream_t stream) {
    const float* x      = (const float*)d_in[0];
    const float* ew1    = (const float*)d_in[1];
    const float* eb1    = (const float*)d_in[2];
    const float* bn1g   = (const float*)d_in[3];
    const float* bn1b   = (const float*)d_in[4];
    const float* ew2    = (const float*)d_in[5];
    const float* eb2    = (const float*)d_in[6];
    const float* bn2g   = (const float*)d_in[7];
    const float* bn2b   = (const float*)d_in[8];
    const float* ew3    = (const float*)d_in[9];
    const float* eb3    = (const float*)d_in[10];
    const float* bn3g   = (const float*)d_in[11];
    const float* bn3b   = (const float*)d_in[12];
    const float* cb     = (const float*)d_in[13];
    const float* dwt    = (const float*)d_in[14];
    const float* dbn1g  = (const float*)d_in[15];
    const float* dbn1b  = (const float*)d_in[16];
    const float* dw2    = (const float*)d_in[17];
    const float* db2    = (const float*)d_in[18];
    const float* dbn2g  = (const float*)d_in[19];
    const float* dbn2b  = (const float*)d_in[20];
    const float* dw3    = (const float*)d_in[21];
    const float* db3    = (const float*)d_in[22];

    float* out   = (float*)d_out;
    float* recon = out;                     // 8*10240
    float* idxf  = out + (size_t)BB * TT;   // 8*2048
    float* loss  = idxf + (size_t)BB * LL;  // 1

    // workspace carve-out (256B aligned)
    char* wsp = (char*)d_ws;
    auto carve = [&](size_t bytes) {
        void* p = (void*)wsp;
        wsp += (bytes + 255) & ~(size_t)255;
        return p;
    };
    float* h1    = (float*)carve(sizeof(float) * (size_t)BB * 16 * TT);
    float* h2    = (float*)carve(sizeof(float) * (size_t)BB * 32 * TT);
    float* z     = (float*)carve(sizeof(float) * (size_t)BB * LL * DD);
    float* q     = (float*)carve(sizeof(float) * (size_t)BB * LL * DD);
    float* hd1   = (float*)carve(sizeof(float) * (size_t)BB * DD * T2);
    float* hd2   = (float*)carve(sizeof(float) * (size_t)BB * DD * T2);
    bf16*  cbb   = (bf16*)carve(sizeof(bf16) * (size_t)KK * DD);
    float* cnorm = (float*)carve(sizeof(float) * KK);
    float* ab    = (float*)carve(sizeof(float) * 128);
    int*   idxi  = (int*)carve(sizeof(int) * (size_t)BB * LL);
    float* lacc  = (float*)carve(sizeof(float) * 4);

    const int TPB = 256;
    auto blocks = [](long n) { return (int)((n + 255) / 256); };

    // codebook prep
    prep_codebook<<<blocks(KK), TPB, 0, stream>>>(cb, cbb, cnorm);

    // encoder
    enc_conv1<<<blocks((long)BB * TT), TPB, 0, stream>>>(x, ew1, eb1, h1);
    bn_stats_nct<<<16, TPB, 0, stream>>>(h1, bn1g, bn1b, ab, 16, TT);
    bn_apply_nct<<<blocks((long)BB * 16 * TT), TPB, 0, stream>>>(h1, ab, 16, TT);
    enc_conv2<<<blocks((long)BB * 32 * TT), TPB, 0, stream>>>(h1, ew2, eb2, h2);
    bn_stats_nct<<<32, TPB, 0, stream>>>(h2, bn2g, bn2b, ab, 32, TT);
    bn_apply_nct<<<blocks((long)BB * 32 * TT), TPB, 0, stream>>>(h2, ab, 32, TT);
    enc_conv3<<<blocks((long)BB * LL * DD), TPB, 0, stream>>>(h2, ew3, eb3, z);
    bn_stats_rc<<<DD, TPB, 0, stream>>>(z, bn3g, bn3b, ab, DD, BB * LL);
    bn_apply_rc<<<blocks((long)BB * LL * DD), TPB, 0, stream>>>(z, ab, DD, BB * LL);

    // VQ nearest neighbor (WMMA + TDM-staged codebook): 1024 row-tiles / 8 waves per block
    vq_argmin<<<(BB * LL / 16) / 8, TPB, 0, stream>>>(z, cbb, cnorm, idxi, idxf);

    // commit loss
    loss_zero<<<1, 1, 0, stream>>>(lacc);
    loss_acc_k<<<blocks((long)BB * LL), TPB, 0, stream>>>(z, cb, idxi, lacc);
    loss_fin<<<1, 1, 0, stream>>>(lacc, loss);

    // decoder
    gather_q<<<blocks((long)BB * LL * DD), TPB, 0, stream>>>(cb, idxi, q);
    dec_tconv1<<<blocks((long)BB * DD * T2), TPB, 0, stream>>>(q, dwt, hd1);
    bn_stats_nct<<<DD, TPB, 0, stream>>>(hd1, dbn1g, dbn1b, ab, DD, T2);
    bn_apply_nct<<<blocks((long)BB * DD * T2), TPB, 0, stream>>>(hd1, ab, DD, T2);
    dec_conv2<<<blocks((long)BB * DD * T2), TPB, 0, stream>>>(hd1, dw2, db2, hd2);
    bn_stats_nct<<<DD, TPB, 0, stream>>>(hd2, dbn2g, dbn2b, ab, DD, T2);
    bn_apply_nct<<<blocks((long)BB * DD * T2), TPB, 0, stream>>>(hd2, ab, DD, T2);
    dec_conv3<<<blocks((long)BB * TT), TPB, 0, stream>>>(hd2, dw3, db3, recon);
}
